// BIOTEncoder_44762149159111
// MI455X (gfx1250) — compile-verified, matrix operand order checked by reading, not run on hardware
//
#include <hip/hip_runtime.h>
#include <hip/hip_bf16.h>
#include <math.h>

// ---------- types ----------
typedef __bf16 bf16_t;
typedef __attribute__((ext_vector_type(16))) __bf16       v16bf;
typedef __attribute__((ext_vector_type(8)))  float        v8f;
typedef __attribute__((ext_vector_type(4)))  unsigned int u32x4;
typedef __attribute__((ext_vector_type(4)))  float        f32x4;

union FragU { u32x4 q[2]; v16bf v; };

enum { EPI_F32 = 0, EPI_RESID = 1, EPI_GELU = 2 };

__device__ __forceinline__ float gelu_tanh(float x) {
    float x3 = x * x * x;
    return 0.5f * x * (1.0f + tanhf(0.7978845608028654f * (x + 0.044715f * x3)));
}

// ---------------------------------------------------------------------------
// WMMA GEMM: C[M,N] = A[M,K](bf16, row-major, ldA) @ B (given as Bt[N,K] bf16
// row-major) (+bias). Block tile 128x128, 8 waves (4x2), wave tile 32x64 =
// 2x4 WMMA frags, BK=32, double-buffered LDS, register-staged prefetch.
// ---------------------------------------------------------------------------
template <int EPI>
__global__ __launch_bounds__(256)
void gemm_bf16_wmma(const bf16_t* __restrict__ A, int ldA,
                    const bf16_t* __restrict__ Bt, int N, int K,
                    const float* __restrict__ bias,
                    void* __restrict__ outv)
{
    // stride 40 bf16 = 80 B rows: 16B-aligned b128 frag reads, conflict-free
    __shared__ bf16_t lA[2][128 * 40];
    __shared__ bf16_t lB[2][128 * 40];

    const int tid    = threadIdx.x;
    const int lane   = tid & 31;
    const int wave   = tid >> 5;
    const int waveM  = wave & 3;   // 0..3
    const int waveN  = wave >> 2;  // 0..1
    const int m0     = blockIdx.x * 128;
    const int n0     = blockIdx.y * 128;
    const int laneLo = lane & 15;
    const int laneHi = (lane >> 4) & 1;

    const int tr = tid >> 1;        // tile row handled by this thread (0..127)
    const int th = tid & 1;         // which 16-elem half of the 32-wide row

    v8f acc[2][4];
    v8f zero = {0.f,0.f,0.f,0.f,0.f,0.f,0.f,0.f};
    #pragma unroll
    for (int mi = 0; mi < 2; ++mi)
        #pragma unroll
        for (int ni = 0; ni < 4; ++ni) acc[mi][ni] = zero;

    const bf16_t* gA = A  + (size_t)(m0 + tr) * ldA + th * 16;
    const bf16_t* gB = Bt + (size_t)(n0 + tr) * K   + th * 16;
    bf16_t* sA[2] = { lA[0] + tr * 40 + th * 16, lA[1] + tr * 40 + th * 16 };
    bf16_t* sB[2] = { lB[0] + tr * 40 + th * 16, lB[1] + tr * 40 + th * 16 };

    // prologue: stage tile 0
    {
        const u32x4* pa = (const u32x4*)gA;
        const u32x4* pb = (const u32x4*)gB;
        u32x4 a0 = pa[0], a1 = pa[1];
        u32x4 b0 = pb[0], b1 = pb[1];
        ((u32x4*)sA[0])[0] = a0; ((u32x4*)sA[0])[1] = a1;
        ((u32x4*)sB[0])[0] = b0; ((u32x4*)sB[0])[1] = b1;
    }
    __syncthreads();

    const int nk = K >> 5;
    for (int ik = 0; ik < nk; ++ik) {
        const int cur = ik & 1;
        const bool more = (ik + 1) < nk;

        // register-staged prefetch of next tile
        u32x4 a0, a1, b0, b1;
        if (more) {
            const u32x4* pa = (const u32x4*)(gA + (size_t)(ik + 1) * 32);
            const u32x4* pb = (const u32x4*)(gB + (size_t)(ik + 1) * 32);
            a0 = pa[0]; a1 = pa[1];
            b0 = pb[0]; b1 = pb[1];
        }

        // fragment gather (ISA 16-bit A layout; B frag = A-layout of Bt)
        FragU af[2], bfg[4];
        #pragma unroll
        for (int mi = 0; mi < 2; ++mi) {
            int r = waveM * 32 + mi * 16 + laneLo;
            const u32x4* p = (const u32x4*)((const char*)lA[cur] + r * 80 + laneHi * 16);
            af[mi].q[0] = p[0];
            af[mi].q[1] = p[2];   // +32 bytes
        }
        #pragma unroll
        for (int ni = 0; ni < 4; ++ni) {
            int c = waveN * 64 + ni * 16 + laneLo;
            const u32x4* p = (const u32x4*)((const char*)lB[cur] + c * 80 + laneHi * 16);
            bfg[ni].q[0] = p[0];
            bfg[ni].q[1] = p[2];
        }

        #pragma unroll
        for (int mi = 0; mi < 2; ++mi)
            #pragma unroll
            for (int ni = 0; ni < 4; ++ni)
                acc[mi][ni] = __builtin_amdgcn_wmma_f32_16x16x32_bf16(
                    false, af[mi].v, false, bfg[ni].v, (short)0, acc[mi][ni], false, false);

        // stage prefetched tile into the alternate LDS buffer
        if (more) {
            int nxt = cur ^ 1;
            ((u32x4*)sA[nxt])[0] = a0; ((u32x4*)sA[nxt])[1] = a1;
            ((u32x4*)sB[nxt])[0] = b0; ((u32x4*)sB[nxt])[1] = b1;
        }
        __syncthreads();
    }

    // epilogue: C frag layout: vgpr r -> row = base + r + (laneHi?8:0), col = base + laneLo
    #pragma unroll
    for (int mi = 0; mi < 2; ++mi) {
        #pragma unroll
        for (int ni = 0; ni < 4; ++ni) {
            int row0 = m0 + waveM * 32 + mi * 16 + laneHi * 8;
            int col  = n0 + waveN * 64 + ni * 16 + laneLo;
            float bv = bias ? bias[col] : 0.0f;
            #pragma unroll
            for (int r = 0; r < 8; ++r) {
                size_t idx = (size_t)(row0 + r) * N + col;
                float v = acc[mi][ni][r] + bv;
                if (EPI == EPI_F32) {
                    ((float*)outv)[idx] = v;
                } else if (EPI == EPI_RESID) {
                    float* o = (float*)outv;
                    o[idx] = o[idx] + v;
                } else { // EPI_GELU
                    ((bf16_t*)outv)[idx] = (bf16_t)gelu_tanh(v);
                }
            }
        }
    }
}

// ---------- prep kernels ----------

// Transposed DFT basis: Bt[256][224]; rows [0,112) = cos(2*pi*j*k/200) (j<101),
// rows [112,224) = sin, rest 0.
__global__ void k_basisT(bf16_t* __restrict__ bt) {
    int n = blockIdx.x, k = threadIdx.x;  // 256 blocks x 224 threads
    float v = 0.0f;
    if (n < 224 && k < 200) {
        int j = (n < 112) ? n : n - 112;
        if (j < 101) {
            float ang = 6.283185307179586f * (float)j * (float)k * (1.0f / 200.0f);
            v = (n < 112) ? cosf(ang) : sinf(ang);
        }
    }
    bt[n * 224 + k] = (bf16_t)v;
}

// dst[N][Kp] (bf16) = transpose of src[srcK][N] (f32), zero-padded rows k>=srcK
__global__ void k_transconv(const float* __restrict__ src, bf16_t* __restrict__ dst,
                            int srcK, int N, int Kp) {
    int i = blockIdx.x * 256 + threadIdx.x;
    if (i < N * Kp) {
        int n = i / Kp, k = i % Kp;
        dst[i] = (bf16_t)(k < srcK ? src[(size_t)k * N + n] : 0.0f);
    }
}

__global__ void k_posenc(float* __restrict__ pe) {
    int t = blockIdx.x, n = threadIdx.x;
    int i = n >> 1;
    float f = __expf(-(float)(2 * i) * 0.03597789207803197f); // ln(10000)/256
    float a = (float)t * f;
    pe[t * 256 + n] = (n & 1) ? cosf(a) : sinf(a);
}

// frames[m][0..224) bf16, m = (b*23+c)*19+t, frame k -> x[b][c][t*100+k]
__global__ void k_frames(const float* __restrict__ x, bf16_t* __restrict__ fr) {
    size_t m = blockIdx.x; int k = threadIdx.x; // 224 threads
    int b = (int)(m / 437), r = (int)(m % 437);
    int c = r / 19, t = r % 19;
    float v = 0.0f;
    if (k < 200) v = x[((size_t)(b * 23 + c)) * 2000 + t * 100 + k];
    fr[m * 224 + k] = (bf16_t)v;
}

__global__ void k_mag(const float* __restrict__ z, bf16_t* __restrict__ spec) {
    size_t m = blockIdx.x; int j = threadIdx.x; // 128 threads
    float v = 0.0f;
    if (j < 112) {
        float re = z[m * 256 + j], im = z[m * 256 + 112 + j];
        v = sqrtf(re * re + im * im);
    }
    spec[m * 128 + j] = (bf16_t)v;
}

// emb = emb_pre + channel_token[c] + posenc[t]; write emb/masked/mask outputs + init h
__global__ void k_finish_emb(const float* __restrict__ ep, const float* __restrict__ ct,
                             const float* __restrict__ pe, const float* __restrict__ noise,
                             float* __restrict__ out_emb, float* __restrict__ out_masked,
                             float* __restrict__ hbuf, float* __restrict__ out_mask) {
    size_t m = blockIdx.x; int n = threadIdx.x;
    int r = (int)(m % 437); int c = r / 19, t = r % 19;
    size_t idx = m * 256 + n;
    float e = ep[idx] + ct[c * 256 + n] + pe[t * 256 + n];
    out_emb[idx] = e;
    bool msk = noise[idx] < 0.9f;
    float mv = msk ? 0.0f : e;
    out_masked[idx] = mv;
    hbuf[idx] = mv;
    out_mask[idx] = msk ? 1.0f : 0.0f;
}

// one wave per row: LN over 256 cols, write bf16
__global__ __launch_bounds__(256)
void k_layernorm(const float* __restrict__ h, const float* __restrict__ s,
                 const float* __restrict__ b, bf16_t* __restrict__ y) {
    int lane = threadIdx.x & 31, wave = threadIdx.x >> 5;
    size_t row = (size_t)blockIdx.x * 8 + wave;
    const float* hr = h + row * 256;
    f32x4 a0 = *(const f32x4*)(hr + lane * 8);
    f32x4 a1 = *(const f32x4*)(hr + lane * 8 + 4);
    float v[8] = {a0.x, a0.y, a0.z, a0.w, a1.x, a1.y, a1.z, a1.w};
    float sum = 0.f;
    #pragma unroll
    for (int i = 0; i < 8; ++i) sum += v[i];
    #pragma unroll
    for (int m = 16; m; m >>= 1) sum += __shfl_xor(sum, m, 32);
    float mu = sum * (1.0f / 256.0f);
    float vs = 0.f;
    #pragma unroll
    for (int i = 0; i < 8; ++i) { float d = v[i] - mu; vs += d * d; }
    #pragma unroll
    for (int m = 16; m; m >>= 1) vs += __shfl_xor(vs, m, 32);
    float inv = rsqrtf(vs * (1.0f / 256.0f) + 1e-5f);
    #pragma unroll
    for (int i = 0; i < 8; ++i) {
        int col = lane * 8 + i;
        y[row * 256 + col] = (bf16_t)((v[i] - mu) * inv * s[col] + b[col]);
    }
}

// fused qkv buffer layout: [M][768] f32, cols 0..255=q, 256..511=k, 512..767=v

// softmax over DHEAD=32 within each head, * 32^-0.5 ; block = token, wave = head
__global__ void k_softmax_q(float* __restrict__ qkv) {
    int lane = threadIdx.x & 31, h = threadIdx.x >> 5;
    size_t idx = (size_t)blockIdx.x * 768 + h * 32 + lane;
    float x = qkv[idx];
    float mx = x;
    #pragma unroll
    for (int m = 16; m; m >>= 1) mx = fmaxf(mx, __shfl_xor(mx, m, 32));
    float e = __expf(x - mx);
    float sm = e;
    #pragma unroll
    for (int m = 16; m; m >>= 1) sm += __shfl_xor(sm, m, 32);
    qkv[idx] = e / sm * 0.17677669529663687f; // 32^-0.5
}

// softmax over sequence (437) per (batch, col); block = batch, thread = col
__global__ void k_softmax_k(float* __restrict__ qkv) {
    int b = blockIdx.x, col = threadIdx.x;
    float* base = qkv + (size_t)b * 437 * 768 + 256 + col;
    float mx = -3.0e38f;
    for (int s = 0; s < 437; ++s) mx = fmaxf(mx, base[(size_t)s * 768]);
    float sum = 0.f;
    for (int s = 0; s < 437; ++s) {
        float e = __expf(base[(size_t)s * 768] - mx);
        base[(size_t)s * 768] = e;
        sum += e;
    }
    float inv = 1.0f / sum;
    for (int s = 0; s < 437; ++s) base[(size_t)s * 768] *= inv;
}

// ctx[b,h,d,e] = sum_s k[b,s,h,d] * v[b,s,h,e]; block = (b,h)
__global__ void k_ctx(const float* __restrict__ qkv, float* __restrict__ ctx) {
    int bh = blockIdx.x, b = bh >> 3, h = bh & 7;
    int e = threadIdx.x & 31, d0 = threadIdx.x >> 5; // d0 0..7
    const float* kp = qkv + (size_t)b * 437 * 768 + 256 + h * 32;
    const float* vp = qkv + (size_t)b * 437 * 768 + 512 + h * 32;
    float acc[4] = {0.f, 0.f, 0.f, 0.f};
    for (int s = 0; s < 437; ++s) {
        float vv = vp[(size_t)s * 768 + e];
        #pragma unroll
        for (int j = 0; j < 4; ++j) acc[j] += kp[(size_t)s * 768 + d0 + j * 8] * vv;
    }
    #pragma unroll
    for (int j = 0; j < 4; ++j)
        ctx[((size_t)bh * 32 + d0 + j * 8) * 32 + e] = acc[j];
}

// att[m,n] = sum_d q[m, h*32+d] * ctx[b,h,d,e]; block = token
__global__ void k_att(const float* __restrict__ qkv, const float* __restrict__ ctx,
                      bf16_t* __restrict__ att) {
    size_t m = blockIdx.x; int n = threadIdx.x;
    int h = n >> 5, e = n & 31;
    int b = (int)(m / 437);
    const float* qp = qkv + m * 768 + h * 32;
    const float* cp = ctx + ((size_t)(b * 8 + h) * 32) * 32 + e;
    float acc = 0.f;
    #pragma unroll 8
    for (int d = 0; d < 32; ++d) acc += qp[d] * cp[d * 32];
    att[m * 256 + n] = (bf16_t)acc;
}

// ---------- launcher ----------
extern "C" void kernel_launch(void* const* d_in, const int* in_sizes, int n_in,
                              void* d_out, int out_size, void* d_ws, size_t ws_size,
                              hipStream_t stream) {
    (void)in_sizes; (void)n_in; (void)out_size; (void)ws_size;

    const float* x      = (const float*)d_in[0];
    const float* noise  = (const float*)d_in[1];
    const float* W_proj = (const float*)d_in[2];
    const float* b_proj = (const float*)d_in[3];
    const float* ct     = (const float*)d_in[4];
    const float* ln1_s  = (const float*)d_in[5];
    const float* ln1_b  = (const float*)d_in[6];
    const float* Wq     = (const float*)d_in[7];
    const float* Wk     = (const float*)d_in[8];
    const float* Wv     = (const float*)d_in[9];
    const float* Wo     = (const float*)d_in[10];
    const float* bo     = (const float*)d_in[11];
    const float* ln2_s  = (const float*)d_in[12];
    const float* ln2_b  = (const float*)d_in[13];
    const float* W1     = (const float*)d_in[14];
    const float* b1     = (const float*)d_in[15];
    const float* W2     = (const float*)d_in[16];
    const float* b2     = (const float*)d_in[17];

    const size_t M = 111872;            // = 256*23*19 = 874*128 exactly
    const size_t SEC = M * 256;
    float* out_emb    = (float*)d_out;
    float* out_masked = out_emb + SEC;
    float* hbuf       = out_emb + 2 * SEC; // final h written in place = output section 2
    float* out_mask   = out_emb + 3 * SEC;

    char* w = (char*)d_ws;
    auto alloc = [&](size_t bytes) { char* p = w; w += (bytes + 255) & ~(size_t)255; return p; };
    bf16_t* frames  = (bf16_t*)alloc(M * 224 * 2);
    bf16_t* basisT  = (bf16_t*)alloc(256 * 224 * 2);
    float*  bigf    = (float*) alloc(M * 256 * 4);      // DFT re/im, then emb_pre
    bf16_t* spec    = (bf16_t*)alloc(M * 128 * 2);
    bf16_t* WprojT  = (bf16_t*)alloc(256 * 128 * 2);
    float*  pe      = (float*) alloc(19 * 256 * 4);
    bf16_t* WqkvT   = (bf16_t*)alloc(4 * 768 * 256 * 2);
    bf16_t* WoT     = (bf16_t*)alloc(4 * 256 * 256 * 2);
    bf16_t* W1T     = (bf16_t*)alloc(4 * 1024 * 256 * 2);
    bf16_t* W2T     = (bf16_t*)alloc(4 * 256 * 1024 * 2);
    bf16_t* ybuf    = (bf16_t*)alloc(M * 256 * 2);      // LN out; reused for att
    float*  qkv     = (float*) alloc(M * 768 * 4);      // fused q|k|v
    float*  ctxb    = (float*) alloc(2048 * 1024 * 4);
    bf16_t* ubuf    = (bf16_t*)qkv;                     // FFN hidden (M*1024 bf16) aliases qkv

    dim3 blk(256);
    const int NW = 256 * 256, NF = 1024 * 256;
    const int gW = (NW + 255) / 256, gF = (NF + 255) / 256;

    // ---- prep: basis, posenc, frames, transposed bf16 weights ----
    k_basisT<<<256, 224, 0, stream>>>(basisT);
    k_posenc<<<19, 256, 0, stream>>>(pe);
    k_frames<<<(int)M, 224, 0, stream>>>(x, frames);
    k_transconv<<<(256 * 128 + 255) / 256, 256, 0, stream>>>(W_proj, WprojT, 101, 256, 128);
    for (int l = 0; l < 4; ++l) {
        k_transconv<<<gW, 256, 0, stream>>>(Wq + l * 65536, WqkvT + l * 196608,           256, 256, 256);
        k_transconv<<<gW, 256, 0, stream>>>(Wk + l * 65536, WqkvT + l * 196608 + 65536,   256, 256, 256);
        k_transconv<<<gW, 256, 0, stream>>>(Wv + l * 65536, WqkvT + l * 196608 + 131072,  256, 256, 256);
        k_transconv<<<gW, 256, 0, stream>>>(Wo + l * 65536, WoT + l * 65536,              256, 256, 256);
        k_transconv<<<gF, 256, 0, stream>>>(W1 + l * 262144, W1T + l * 262144,            256, 1024, 256);
        k_transconv<<<gF, 256, 0, stream>>>(W2 + l * 262144, W2T + l * 262144,            1024, 256, 1024);
    }

    // ---- STFT magnitude via WMMA GEMM against cos|sin basis ----
    gemm_bf16_wmma<EPI_F32><<<dim3(874, 2), blk, 0, stream>>>(frames, 224, basisT, 256, 224, nullptr, bigf);
    k_mag<<<(int)M, 128, 0, stream>>>(bigf, spec);

    // ---- projection + embedding assembly ----
    gemm_bf16_wmma<EPI_F32><<<dim3(874, 2), blk, 0, stream>>>(spec, 128, WprojT, 256, 128, b_proj, bigf);
    k_finish_emb<<<(int)M, 256, 0, stream>>>(bigf, ct, pe, noise, out_emb, out_masked, hbuf, out_mask);

    // ---- 4 transformer layers (linear attention) ----
    for (int l = 0; l < 4; ++l) {
        k_layernorm<<<(int)(M / 8), 256, 0, stream>>>(hbuf, ln1_s + l * 256, ln1_b + l * 256, ybuf);
        gemm_bf16_wmma<EPI_F32><<<dim3(874, 6), blk, 0, stream>>>(ybuf, 256, WqkvT + l * 196608, 768, 256, nullptr, qkv);
        k_softmax_q<<<(int)M, 256, 0, stream>>>(qkv);
        k_softmax_k<<<256, 256, 0, stream>>>(qkv);
        k_ctx<<<2048, 256, 0, stream>>>(qkv, ctxb);
        k_att<<<(int)M, 256, 0, stream>>>(qkv, ctxb, ybuf);   // att -> ybuf (bf16)
        gemm_bf16_wmma<EPI_RESID><<<dim3(874, 2), blk, 0, stream>>>(ybuf, 256, WoT + l * 65536, 256, 256, bo + l * 256, hbuf);
        k_layernorm<<<(int)(M / 8), 256, 0, stream>>>(hbuf, ln2_s + l * 256, ln2_b + l * 256, ybuf);
        gemm_bf16_wmma<EPI_GELU><<<dim3(874, 8), blk, 0, stream>>>(ybuf, 256, W1T + l * 262144, 1024, 256, b1 + l * 1024, ubuf);
        gemm_bf16_wmma<EPI_RESID><<<dim3(874, 2), blk, 0, stream>>>(ubuf, 1024, W2T + l * 262144, 256, 1024, b2 + l * 256, hbuf);
    }
    // final h already resides in d_out section 2
}
